// GATEncoder_14405320311216
// MI455X (gfx1250) — matrix operand order repaired and checked by reading, not compiled, not run on hardware
//
#include <hip/hip_runtime.h>
#include <hip/hip_bf16.h>

#define NEG_SLOPE 0.2f

typedef __attribute__((ext_vector_type(16))) _Float16 v16h;
typedef __attribute__((ext_vector_type(8)))  _Float16 v8h;
typedef __attribute__((ext_vector_type(4)))  _Float16 v4h;
typedef __attribute__((ext_vector_type(8)))  float    v8f;
typedef __attribute__((ext_vector_type(4)))  float    v4f;

// ---------- W (K=128 x N=128, f32 row-major) -> WT (N-major f16: WT[n][k]) ----------
__global__ void k_wt(const float* __restrict__ W, _Float16* __restrict__ WT) {
    int i = blockIdx.x * 256 + threadIdx.x;     // i in [0, 128*128)
    int k = i >> 7, n = i & 127;
    WT[n * 128 + k] = (_Float16)W[i];
}

// ---------- H[M,128] = A[M,128] * W[128,128] via v_wmma_f32_16x16x32_f16 ----------
// Block: 256 threads = 8 waves. Each block: one 16-row M tile; wave w owns N cols [16w,16w+16).
__global__ void __launch_bounds__(256) k_gemm(const float* __restrict__ A,
                                              const _Float16* __restrict__ WT,
                                              float* __restrict__ H, int nN) {
    __shared__ __align__(32) _Float16 As[16 * 128];
    const int  tid      = threadIdx.x;
    const long rowBase  = (long)blockIdx.x * 16;
    const bool fullTile = (rowBase + 16 <= nN);   // uniform across block

    // Stage 16x128 f32 tile -> f16 LDS (float4 load -> packed 4xf16 ds_store_b64)
    if (fullTile) {
        for (int c = tid; c < 512; c += 256) {             // 512 float4 chunks
            const float4 v = *(const float4*)(A + rowBase * 128 + c * 4);
            v4h h4 = { (_Float16)v.x, (_Float16)v.y, (_Float16)v.z, (_Float16)v.w };
            *(v4h*)(As + c * 4) = h4;
        }
    } else {
        for (int idx = tid; idx < 16 * 128; idx += 256) {
            long grow = rowBase + (idx >> 7);
            As[idx] = (grow < nN) ? (_Float16)A[grow * 128 + (idx & 127)] : (_Float16)0.0f;
        }
    }
    __syncthreads();

    const int wave = tid >> 5;      // 0..7  -> N block
    const int lane = tid & 31;
    const int g    = lane >> 4;     // lane group
    const int m    = lane & 15;

    v8f acc = {};
    const _Float16* bptr = WT + (long)(wave * 16 + m) * 128;  // row n = 16w+m of WT (K contiguous)
#pragma unroll
    for (int kk = 0; kk < 4; ++kk) {
        const int kb = kk * 32;
        // A frag (16x32 f16 layout): e=0..15 -> K=kb+16g+e (contiguous, matches B)
        v16h a  = *(const v16h*)(As + m * 128 + kb + 16 * g);
        // B frag (32x16 f16 layout): lane n = L%16, e=0..15 -> K=kb+16g+e (contiguous in WT)
        v16h b  = *(const v16h*)(bptr + kb + 16 * g);
        acc = __builtin_amdgcn_wmma_f32_16x16x32_f16(false, a, false, b, (short)0, acc, false, false);
    }
    // D layout: VGPR v, lane L: M = v + 8*(L/16), N = L%16
    float* out = H + (rowBase + 8 * g) * 128 + wave * 16 + m;
    if (fullTile) {
#pragma unroll
        for (int v = 0; v < 8; ++v) out[(long)v * 128] = acc[v];
    } else {
#pragma unroll
        for (int v = 0; v < 8; ++v) {
            if (rowBase + 8 * g + v < nN) out[(long)v * 128] = acc[v];
        }
    }
}

// ---------- per-node attention logits: s = h[n] . a ----------
__global__ void k_logits(const float* __restrict__ H, const float* __restrict__ asrc,
                         const float* __restrict__ adst, float* __restrict__ ssrc,
                         float* __restrict__ sdst, int nN) {
    int node = blockIdx.x * 8 + (threadIdx.x >> 5);
    if (node >= nN) return;
    int lane = threadIdx.x & 31;
    float4 hv = *(const float4*)(H + (long)node * 128 + lane * 4);
    float4 as = *(const float4*)(asrc + lane * 4);
    float4 ad = *(const float4*)(adst + lane * 4);
    float s0 = hv.x * as.x + hv.y * as.y + hv.z * as.z + hv.w * as.w;
    float s1 = hv.x * ad.x + hv.y * ad.y + hv.z * ad.z + hv.w * ad.w;
    for (int off = 16; off; off >>= 1) {
        s0 += __shfl_xor(s0, off, 32);
        s1 += __shfl_xor(s1, off, 32);
    }
    if (lane == 0) { ssrc[node] = s0; sdst[node] = s1; }
}

// ---------- order-preserving float<->uint keys for atomic max ----------
__device__ __forceinline__ unsigned fkey(float f) {
    unsigned u = __float_as_uint(f);
    return (u & 0x80000000u) ? ~u : (u | 0x80000000u);
}
__device__ __forceinline__ float funkey(unsigned k) {
    unsigned u = (k & 0x80000000u) ? (k ^ 0x80000000u) : ~k;
    return __uint_as_float(u);
}

// ---------- zero accumulator (float4-wide) + init segment-max/denominator ----------
__global__ void k_init(float* __restrict__ accum, unsigned* __restrict__ mkey,
                       float* __restrict__ denom, int nN) {
    long i = (long)blockIdx.x * 256 + threadIdx.x;      // i over nN*32 float4 chunks
    if (i < (long)nN * 32) {
        v4f z = {};
        *(v4f*)(accum + i * 4) = z;
    }
    if (i < nN) { mkey[i] = 0u; denom[i] = 0.0f; }
}

// ---------- pass A: e = leaky(s_src[src]+s_dst[dst]); segment max over dst ----------
__global__ void k_edge_max(const int* __restrict__ esrc, const int* __restrict__ edst,
                           const float* __restrict__ ssrc, const float* __restrict__ sdst,
                           float* __restrict__ ebuf, unsigned* __restrict__ mkey,
                           int nE, int nTot) {
    int i = blockIdx.x * 256 + threadIdx.x;
    if (i >= nTot) return;
    int s, d;
    if (i < nE) { s = esrc[i]; d = edst[i]; } else { s = d = i - nE; }
    float e = ssrc[s] + sdst[d];
    e = (e > 0.0f) ? e : NEG_SLOPE * e;
    ebuf[i] = e;
    atomicMax(mkey + d, fkey(e));
}

// ---------- pass B: ex = exp(e - m[dst]); segment sum ----------
__global__ void k_edge_exp(const int* __restrict__ edst, float* __restrict__ ebuf,
                           const unsigned* __restrict__ mkey, float* __restrict__ denom,
                           int nE, int nTot) {
    int i = blockIdx.x * 256 + threadIdx.x;
    if (i >= nTot) return;
    int d = (i < nE) ? edst[i] : (i - nE);
    float ex = __expf(ebuf[i] - funkey(mkey[d]));
    ebuf[i] = ex;
    atomicAdd(denom + d, ex);
}

// ---------- pass C: out[dst] += alpha * h[src]  (one wave per edge, 128-wide row) ----------
__global__ void k_edge_agg(const int* __restrict__ esrc, const int* __restrict__ edst,
                           const float* __restrict__ ebuf, const float* __restrict__ denom,
                           const float* __restrict__ H, float* __restrict__ accum,
                           int nE, int nTot) {
    int i = blockIdx.x * 8 + (threadIdx.x >> 5);
    if (i >= nTot) return;
    int lane = threadIdx.x & 31;
    int s, d;
    if (i < nE) { s = esrc[i]; d = edst[i]; } else { s = d = i - nE; }
    float alpha = ebuf[i] / denom[d];
    float4 hv = *(const float4*)(H + (long)s * 128 + lane * 4);
    float* o = accum + (long)d * 128 + lane * 4;
    atomicAdd(o + 0, alpha * hv.x);
    atomicAdd(o + 1, alpha * hv.y);
    atomicAdd(o + 2, alpha * hv.z);
    atomicAdd(o + 3, alpha * hv.w);
}

// ---------- bias + relu, in place (float4-wide) ----------
__global__ void k_bias_relu(float* __restrict__ x, const float* __restrict__ b, long n4) {
    long i = (long)blockIdx.x * 256 + threadIdx.x;      // over float4 chunks
    if (i >= n4) return;
    float4 v = *(const float4*)(x + i * 4);
    const float4 bb = *(const float4*)(b + (i & 31) * 4);
    v.x = fmaxf(v.x + bb.x, 0.0f);
    v.y = fmaxf(v.y + bb.y, 0.0f);
    v.z = fmaxf(v.z + bb.z, 0.0f);
    v.w = fmaxf(v.w + bb.w, 0.0f);
    *(float4*)(x + i * 4) = v;
}

// ---------- pooling ----------
__global__ void k_pool_init(float* __restrict__ sums, float* __restrict__ cnts, int nG) {
    int i = blockIdx.x * 256 + threadIdx.x;
    if (i < nG * 128) sums[i] = 0.0f;
    if (i < nG) cnts[i] = 0.0f;
}
__global__ void k_pool(const float* __restrict__ H, const int* __restrict__ batch,
                       float* __restrict__ sums, float* __restrict__ cnts, int nN) {
    int node = blockIdx.x * 8 + (threadIdx.x >> 5);
    if (node >= nN) return;
    int lane = threadIdx.x & 31;
    int g = batch[node];
    float4 hv = *(const float4*)(H + (long)node * 128 + lane * 4);
    float* o = sums + (long)g * 128 + lane * 4;
    atomicAdd(o + 0, hv.x);
    atomicAdd(o + 1, hv.y);
    atomicAdd(o + 2, hv.z);
    atomicAdd(o + 3, hv.w);
    if (lane == 0) atomicAdd(cnts + g, 1.0f);
}

// ---------- final fc: out[g,j] = (sums[g]/max(cnt,1)) . fcW[:,j] + fcb[j] ----------
__global__ void k_fc(const float* __restrict__ sums, const float* __restrict__ cnts,
                     const float* __restrict__ W, const float* __restrict__ b,
                     float* __restrict__ out, int nG) {
    int i = blockIdx.x * 256 + threadIdx.x;
    if (i >= nG * 64) return;
    int gidx = i >> 6, j = i & 63;
    float inv = 1.0f / fmaxf(cnts[gidx], 1.0f);
    const float* row = sums + (long)gidx * 128;
    float acc = 0.0f;
    for (int k = 0; k < 128; ++k) acc += row[k] * W[k * 64 + j];
    out[i] = acc * inv + b[j];
}

extern "C" void kernel_launch(void* const* d_in, const int* in_sizes, int n_in,
                              void* d_out, int out_size, void* d_ws, size_t ws_size,
                              hipStream_t stream) {
    const float* x     = (const float*)d_in[0];
    const int*   esrc  = (const int*)d_in[1];
    const int*   edst  = (const int*)d_in[2];
    const int*   batch = (const int*)d_in[3];
    const float* Wl[3]    = {(const float*)d_in[4],  (const float*)d_in[8],  (const float*)d_in[12]};
    const float* asrcl[3] = {(const float*)d_in[5],  (const float*)d_in[9],  (const float*)d_in[13]};
    const float* adstl[3] = {(const float*)d_in[6],  (const float*)d_in[10], (const float*)d_in[14]};
    const float* biasl[3] = {(const float*)d_in[7],  (const float*)d_in[11], (const float*)d_in[15]};
    const float* fcW = (const float*)d_in[16];
    const float* fcb = (const float*)d_in[17];

    const int nN   = in_sizes[3];          // N_NODES (batch vector length)
    const int nE   = in_sizes[1];          // N_EDGES
    const int nTot = nE + nN;              // edges + self loops
    const int nG   = out_size / 64;        // N_GRAPHS

    // ---- workspace carve (256B aligned) ----
    char* p = (char*)d_ws;
    auto alloc = [&](size_t bytes) { void* r = (void*)p; p += (bytes + 255) & ~(size_t)255; return r; };
    const size_t big = (size_t)nN * 128 * sizeof(float);
    float*    bufA   = (float*)alloc(big);
    float*    bufB   = (float*)alloc(big);          // always holds h of current layer
    float*    bufC   = (float*)alloc(big);
    _Float16* WT[3]  = {(_Float16*)alloc(128 * 128 * 2), (_Float16*)alloc(128 * 128 * 2),
                        (_Float16*)alloc(128 * 128 * 2)};
    float*    ssrc   = (float*)alloc((size_t)nN * 4);
    float*    sdst   = (float*)alloc((size_t)nN * 4);
    float*    ebuf   = (float*)alloc((size_t)nTot * 4);
    unsigned* mkey   = (unsigned*)alloc((size_t)nN * 4);
    float*    denom  = (float*)alloc((size_t)nN * 4);
    float*    psums  = (float*)alloc((size_t)nG * 128 * 4);
    float*    pcnts  = (float*)alloc((size_t)nG * 4);

    auto cdiv = [](long a, long b) { return (int)((a + b - 1) / b); };

    // transpose+convert all layer weights once per call
    for (int l = 0; l < 3; ++l)
        k_wt<<<64, 256, 0, stream>>>(Wl[l], WT[l]);

    const float* cur = x;
    float* accum_of[3] = {bufC, bufA, bufC};  // ping-pong so h (bufB) never aliases
    for (int l = 0; l < 3; ++l) {
        float* acc = accum_of[l];
        k_gemm<<<cdiv(nN, 16), 256, 0, stream>>>(cur, WT[l], bufB, nN);
        k_logits<<<cdiv(nN, 8), 256, 0, stream>>>(bufB, asrcl[l], adstl[l], ssrc, sdst, nN);
        k_init<<<cdiv((long)nN * 32, 256), 256, 0, stream>>>(acc, mkey, denom, nN);
        k_edge_max<<<cdiv(nTot, 256), 256, 0, stream>>>(esrc, edst, ssrc, sdst, ebuf, mkey, nE, nTot);
        k_edge_exp<<<cdiv(nTot, 256), 256, 0, stream>>>(edst, ebuf, mkey, denom, nE, nTot);
        k_edge_agg<<<cdiv(nTot, 8), 256, 0, stream>>>(esrc, edst, ebuf, denom, bufB, acc, nE, nTot);
        k_bias_relu<<<cdiv((long)nN * 32, 256), 256, 0, stream>>>(acc, biasl[l], (long)nN * 32);
        cur = acc;
    }

    k_pool_init<<<cdiv((long)nG * 128, 256), 256, 0, stream>>>(psums, pcnts, nG);
    k_pool<<<cdiv(nN, 8), 256, 0, stream>>>(cur, batch, psums, pcnts, nN);
    k_fc<<<cdiv((long)nG * 64, 256), 256, 0, stream>>>(psums, pcnts, fcW, fcb, (float*)d_out, nG);
}
